// DualGatedAdaptiveFusionV3_5420248728205
// MI455X (gfx1250) — matrix operand order repaired and checked by reading, not compiled, not run on hardware
//
#include <hip/hip_runtime.h>
#include <hip/hip_bf16.h>
#include <math.h>

// Problem constants (match reference)
#define BATCH 512
#define LTOK  129
#define CDIM  512
#define NHEAD 8
#define DH    64
#define KPOOL (NHEAD * CDIM)   // 4096

typedef _Float16 v16h __attribute__((ext_vector_type(16)));
typedef float    v8f  __attribute__((ext_vector_type(8)));

// ---------------- wave32 / block reductions ----------------
__device__ __forceinline__ float warpSum(float v) {
#pragma unroll
    for (int o = 16; o; o >>= 1) v += __shfl_xor(v, o, 32);
    return v;
}

// blockDim.x == 256 (8 waves of 32)
__device__ float blockSum(float v, float* sm) {
    const int lane = threadIdx.x & 31, wid = threadIdx.x >> 5;
    v = warpSum(v);
    if (lane == 0) sm[wid] = v;
    __syncthreads();
    float r = (threadIdx.x < 8) ? sm[threadIdx.x] : 0.0f;
    if (wid == 0) {
#pragma unroll
        for (int o = 4; o; o >>= 1) r += __shfl_xor(r, o, 32);
        if (lane == 0) sm[0] = r;
    }
    __syncthreads();
    r = sm[0];
    __syncthreads();
    return r;
}

// ---------------- K1: fold query into K-proj; ent_w -> mean row; ebias -----
// qk[m][h][c] = sum_d q_m[h*64+d] * Wk[h*64+d][c],  q_m = Wq @ query_m + bq
// sb[m][h]    = sum_d q_m[h*64+d] * bk[h*64+d]
// emean[c]    = mean_j ent_w[j][c];  emean[512] = mean(ent_b)
// ebias[n]    = out_b[n] + sum_i out_w[n][i] * bv[i]
__global__ __launch_bounds__(256) void precompute_kernel(
    const float* __restrict__ q_rgb, const float* __restrict__ q_nir,
    const float* __restrict__ q_tir,
    const float* __restrict__ in_w, const float* __restrict__ in_b,
    const float* __restrict__ ent_w, const float* __restrict__ ent_b,
    const float* __restrict__ out_w, const float* __restrict__ out_b,
    float* __restrict__ qk, float* __restrict__ sb, float* __restrict__ emean,
    float* __restrict__ ebias)
{
    __shared__ float qs[3][CDIM];
    const int tid = threadIdx.x;
    for (int idx = tid; idx < 3 * CDIM; idx += 256) {
        const int m = idx >> 9, i = idx & 511;
        const float* qv = (m == 0) ? q_rgb : (m == 1) ? q_nir : q_tir;
        const float* wr = in_w + (size_t)i * CDIM;   // Wq row i
        float acc = in_b[i];                          // bq
        for (int c = 0; c < CDIM; ++c) acc = fmaf(wr[c], qv[c], acc);
        qs[m][i] = acc;
    }
    __syncthreads();
    for (int idx = tid; idx < 3 * NHEAD * CDIM; idx += 256) {
        const int m = idx >> 12, h = (idx >> 9) & 7, c = idx & 511;
        float acc = 0.0f;
        for (int d = 0; d < DH; ++d)
            acc = fmaf(qs[m][h * DH + d],
                       in_w[(size_t)(CDIM + h * DH + d) * CDIM + c], acc);
        qk[idx] = acc;
    }
    if (tid < 24) {
        const int m = tid / 8, h = tid % 8;
        float acc = 0.0f;
        for (int d = 0; d < DH; ++d)
            acc = fmaf(qs[m][h * DH + d], in_b[CDIM + h * DH + d], acc);
        sb[tid] = acc;
    }
    for (int c = tid; c < CDIM; c += 256) {
        float acc = 0.0f;
        for (int j = 0; j < CDIM; ++j) acc += ent_w[(size_t)j * CDIM + c];
        emean[c] = acc * (1.0f / CDIM);
    }
    if (tid == 0) {
        float acc = 0.0f;
        for (int j = 0; j < CDIM; ++j) acc += ent_b[j];
        emean[CDIM] = acc * (1.0f / CDIM);
    }
    for (int n = tid; n < CDIM; n += 256) {
        const float* wr = out_w + (size_t)n * CDIM;
        const float* bv = in_b + 2 * CDIM;
        float acc = out_b[n];
        for (int i = 0; i < CDIM; ++i) acc = fmaf(wr[i], bv[i], acc);
        ebias[n] = acc;
    }
}

// ---------------- K1b: Weff[n][h*512+c] = sum_d Wout[n][h*64+d]*Wv[h*64+d][c]
// grid (NHEAD, CDIM/128); Wv chunk staged in LDS, reused for all 512 n.
__global__ __launch_bounds__(256) void weff_kernel(
    const float* __restrict__ out_w, const float* __restrict__ in_w,
    float* __restrict__ weff)
{
    __shared__ float Vs[DH][128];
    const int h = blockIdx.x, c0 = blockIdx.y * 128;
    const int tid = threadIdx.x;
    const float* Wv = in_w + (size_t)(2 * CDIM) * CDIM;
#pragma unroll
    for (int i = 0; i < 8; ++i) {               // 2048 float4 chunks
        const int ch = tid + (i << 8);
        const int d = ch >> 5, cc = (ch & 31) << 2;
        const float4 v = *(const float4*)(Wv + (size_t)(h * DH + d) * CDIM + c0 + cc);
        Vs[d][cc] = v.x; Vs[d][cc + 1] = v.y; Vs[d][cc + 2] = v.z; Vs[d][cc + 3] = v.w;
    }
    __syncthreads();
    const int cl = tid & 127;                   // local column
    const int nh = tid >> 7;                    // 0..1
    for (int n0 = 0; n0 < CDIM; n0 += 2) {
        const int n = n0 + nh;
        const float* wr = out_w + (size_t)n * CDIM + h * DH;
        float acc = 0.0f;
#pragma unroll
        for (int d = 0; d < DH; ++d) acc = fmaf(wr[d], Vs[d][cl], acc);
        weff[(size_t)n * KPOOL + h * CDIM + c0 + cl] = acc;
    }
}

// ---------------- K2: single-pass online-softmax attention pooling ----------
// One 256-thread block per (b, modality). Tokens streamed from HBM once.
// wave w owns head w: running (max, denom) + pooled[w][0..511] in LDS.
// Writes normalized pooled vectors (coalesced) -> pool[(m*B+b)][h*512+c].
__global__ __launch_bounds__(256) void attn_pool_kernel(
    const float* __restrict__ rgb, const float* __restrict__ nir,
    const float* __restrict__ tir,
    const float* __restrict__ qk, const float* __restrict__ sb,
    float* __restrict__ pool)
{
    __shared__ float tok[CDIM];
    __shared__ float pooled[NHEAD][CDIM];
    const int b = blockIdx.x, m = blockIdx.y;
    const float* tptr = (m == 0) ? rgb : (m == 1) ? nir : tir;
    const float* src = tptr + (size_t)b * LTOK * CDIM;
    const int tid = threadIdx.x, lane = tid & 31, wid = tid >> 5;

    for (int i = tid; i < NHEAD * CDIM; i += 256) ((float*)pooled)[i] = 0.0f;

    // stage this wave's folded-query row in registers (16 floats per lane)
    const float* qkrow = qk + ((size_t)m * NHEAD + wid) * CDIM;
    float qr[16];
#pragma unroll
    for (int j = 0; j < 16; ++j) qr[j] = qkrow[lane + 32 * j];
    const float sbias = sb[m * NHEAD + wid];

    float mrun = -INFINITY, drun = 0.0f;
    for (int l = 0; l < LTOK; ++l) {
        __syncthreads();
        {
            const float2 v = *(const float2*)(src + (size_t)l * CDIM + tid * 2);
            tok[tid * 2] = v.x; tok[tid * 2 + 1] = v.y;
            if (l + 1 < LTOK)   // global_prefetch_b8 next token line
                __builtin_prefetch(src + (size_t)(l + 1) * CDIM + tid * 2, 0, 0);
        }
        __syncthreads();
        float part = 0.0f;
#pragma unroll
        for (int j = 0; j < 16; ++j) part = fmaf(tok[lane + 32 * j], qr[j], part);
        const float s = (warpSum(part) + sbias) * 0.125f;  // *1/sqrt(64)
        const float mnew = fmaxf(mrun, s);
        const float fac = __expf(mrun - mnew);   // 0 on first token
        const float p   = __expf(s - mnew);
        drun = drun * fac + p;
        mrun = mnew;
#pragma unroll
        for (int j = 0; j < 16; ++j) {
            const int c = lane + 32 * j;
            pooled[wid][c] = pooled[wid][c] * fac + p * tok[c];
        }
    }
    const float inv = 1.0f / drun;
    float* dst = pool + (size_t)(m * BATCH + b) * KPOOL + wid * CDIM;
#pragma unroll
    for (int j = 0; j < 16; ++j) {
        const int c = lane + 32 * j;
        dst[c] = pooled[wid][c] * inv;
    }
}

// ---------------- generic WMMA GEMM: D = A(MxK) * W(NxK)^T + bias ----------
// 256 threads = 8 waves; block tile 64(M) x 128(N); K stepped by 32.
// Wave w owns N-strip w*16; 4 accumulators over M sub-tiles -> 4 WMMA/step
// sharing one B fragment. Fragments per ISA §7.12.2 16-bit layouts.
__global__ __launch_bounds__(256) void gemm_wmma(
    const float* __restrict__ A, const float* __restrict__ W,
    const float* __restrict__ bias, float* __restrict__ D,
    int M, int N, int K)
{
    __shared__ _Float16 As[64][32];    // [m][k]
    __shared__ _Float16 Bs[128][32];   // [n][k]  (table-order B fragments)
    const int tid = threadIdx.x;
    const int lane = tid & 31, wid = tid >> 5;
    const int hi = lane >> 4, ln = lane & 15;
    const int m0 = blockIdx.x * 64;
    const int n0 = blockIdx.y * 128;
    v8f acc[4] = {};
    for (int k0 = 0; k0 < K; k0 += 32) {
        __syncthreads();
#pragma unroll
        for (int i = 0; i < 2; ++i) {            // A slab 64x32 (512 float4)
            const int ch = tid + (i << 8);
            const int r = ch >> 3, kc = (ch & 7) << 2;
            const float4 va = *(const float4*)(A + (size_t)(m0 + r) * K + k0 + kc);
            As[r][kc + 0] = (_Float16)va.x; As[r][kc + 1] = (_Float16)va.y;
            As[r][kc + 2] = (_Float16)va.z; As[r][kc + 3] = (_Float16)va.w;
        }
#pragma unroll
        for (int i = 0; i < 4; ++i) {            // B slab 128x32 (1024 float4)
            const int ch = tid + (i << 8);
            const int nn = ch >> 3, kq = (ch & 7) << 2;
            const float4 vb = *(const float4*)(W + (size_t)(n0 + nn) * K + k0 + kq);
            Bs[nn][kq + 0] = (_Float16)vb.x; Bs[nn][kq + 1] = (_Float16)vb.y;
            Bs[nn][kq + 2] = (_Float16)vb.z; Bs[nn][kq + 3] = (_Float16)vb.w;
        }
        __syncthreads();
        // B 32x16 fragment: e0..7 -> K = 8*hi + e ; e8..15 -> K = 16 + 8*hi + (e-8)
        v16h bf;
#pragma unroll
        for (int e = 0; e < 16; ++e) bf[e] = Bs[wid * 16 + ln][8 * hi + e + ((e >= 8) ? 8 : 0)];
#pragma unroll
        for (int s = 0; s < 4; ++s) {
            // A 16x32 fragment: v0..3 -> K = 8*hi + 2v+hl ; v4..7 -> +16
            v16h af;
#pragma unroll
            for (int e = 0; e < 16; ++e) {
                const int v = e >> 1, hl = e & 1;
                const int ka = (v < 4) ? (8 * hi + 2 * v + hl)
                                       : (16 + 8 * hi + 2 * (v - 4) + hl);
                af[e] = As[s * 16 + ln][ka];
            }
            acc[s] = __builtin_amdgcn_wmma_f32_16x16x32_f16(
                false, af, false, bf, (short)0, acc[s], false, false);
        }
    }
    const int col = n0 + wid * 16 + ln;
    const float bcol = bias[col];
#pragma unroll
    for (int s = 0; s < 4; ++s)
#pragma unroll
        for (int r = 0; r < 8; ++r) {            // C/D: vgpr r -> M = r + 8*hi
            const int mrow = m0 + s * 16 + r + 8 * hi;
            D[(size_t)mrow * N + col] = acc[s][r] + bcol;
        }
}

// ---------------- row LayerNorm (optionally + ReLU), rows of 512 ----------
__global__ __launch_bounds__(256) void ln_rows_kernel(
    const float* __restrict__ X, const float* __restrict__ g,
    const float* __restrict__ b, float* __restrict__ Y, int applyRelu)
{
    __shared__ float sm[8];
    const int row = blockIdx.x, tid = threadIdx.x;
    const float* x = X + (size_t)row * CDIM;
    const float x0 = x[tid], x1 = x[tid + 256];
    const float s  = blockSum(x0 + x1, sm);
    const float sq = blockSum(x0 * x0 + x1 * x1, sm);
    const float mean = s * (1.0f / CDIM);
    const float var  = sq * (1.0f / CDIM) - mean * mean;
    const float rstd = rsqrtf(var + 1e-5f);
    float y0 = (x0 - mean) * rstd * g[tid] + b[tid];
    float y1 = (x1 - mean) * rstd * g[tid + 256] + b[tid + 256];
    if (applyRelu) { y0 = fmaxf(y0, 0.0f); y1 = fmaxf(y1, 0.0f); }
    float* y = Y + (size_t)row * CDIM;
    y[tid] = y0; y[tid + 256] = y1;
}

// ---------------- pack h (3,B,C) -> hc (B,3C) ------------------------------
__global__ __launch_bounds__(256) void pack_hc(const float* __restrict__ h,
                                               float* __restrict__ hc)
{
    const int idx = blockIdx.x * 256 + threadIdx.x;     // < B*3*C
    const int b = idx / (3 * CDIM), r = idx % (3 * CDIM);
    const int m = r >> 9, c = r & 511;
    hc[idx] = h[((size_t)m * BATCH + b) * CDIM + c];
}

// ---------------- per-batch fusion: entropy weights + gates + blend --------
__global__ __launch_bounds__(256) void fuse_kernel(
    const float* __restrict__ h, const float* __restrict__ emean,
    const float* __restrict__ g, const float* __restrict__ w2,
    const float* __restrict__ b2, const float* __restrict__ alpha_raw,
    float* __restrict__ hfus)
{
    __shared__ float sm[8];
    const int b = blockIdx.x, tid = threadIdx.x;
    float xs[3][2];
    float score[3];
    for (int m = 0; m < 3; ++m) {
        const float* x = h + ((size_t)m * BATCH + b) * CDIM;
        const float x0 = x[tid], x1 = x[tid + 256];
        xs[m][0] = x0; xs[m][1] = x1;
        const float fa0 = fabsf(x0) + 1e-8f, fa1 = fabsf(x1) + 1e-8f;
        const float S = blockSum(fa0 + fa1, sm);
        const float invS = 1.0f / S;
        const float p0 = fa0 * invS, p1 = fa1 * invS;
        const float H = -blockSum(p0 * logf(p0 + 1e-8f) + p1 * logf(p1 + 1e-8f), sm);
        const float z = blockSum(x0 * emean[tid] + x1 * emean[tid + 256], sm)
                        + emean[CDIM];
        score[m] = z * expf(-H);                 // TAU = 1
    }
    const float smax = fmaxf(score[0], fmaxf(score[1], score[2]));
    const float e0 = expf(score[0] - smax), e1 = expf(score[1] - smax),
                e2 = expf(score[2] - smax);
    const float einv = 1.0f / (e0 + e1 + e2);
    const float ew0 = e0 * einv, ew1 = e1 * einv, ew2 = e2 * einv;

    const float* grow = g + (size_t)b * CDIM;
    const float g0v = grow[tid], g1v = grow[tid + 256];
    float gate[3];
    for (int j = 0; j < 3; ++j) {
        const float dot = blockSum(g0v * w2[j * CDIM + tid]
                                 + g1v * w2[j * CDIM + tid + 256], sm) + b2[j];
        gate[j] = 1.0f / (1.0f + expf(-dot));
    }
    const float alpha = 1.0f / (1.0f + expf(-alpha_raw[0]));
    float* out = hfus + (size_t)b * CDIM;
#pragma unroll
    for (int half = 0; half < 2; ++half) {
        const int c = tid + half * 256;
        const float he = ew0 * xs[0][half] + ew1 * xs[1][half] + ew2 * xs[2][half];
        const float hi = gate[0] * xs[0][half] + gate[1] * xs[1][half]
                       + gate[2] * xs[2][half];
        out[c] = alpha * he + (1.0f - alpha) * hi;
    }
}

// ---------------- final: LN(enh) then out[b] = concat(h_m[b] + h_enh[b]) ---
__global__ __launch_bounds__(256) void final_kernel(
    const float* __restrict__ elin, const float* __restrict__ g,
    const float* __restrict__ b, const float* __restrict__ h,
    float* __restrict__ out)
{
    __shared__ float sm[8];
    const int row = blockIdx.x, tid = threadIdx.x;
    const float* x = elin + (size_t)row * CDIM;
    const float x0 = x[tid], x1 = x[tid + 256];
    const float s  = blockSum(x0 + x1, sm);
    const float sq = blockSum(x0 * x0 + x1 * x1, sm);
    const float mean = s * (1.0f / CDIM);
    const float var  = sq * (1.0f / CDIM) - mean * mean;
    const float rstd = rsqrtf(var + 1e-5f);
    const float e0 = (x0 - mean) * rstd * g[tid] + b[tid];
    const float e1 = (x1 - mean) * rstd * g[tid + 256] + b[tid + 256];
    float* orow = out + (size_t)row * 3 * CDIM;
    for (int m = 0; m < 3; ++m) {
        const float* hrow = h + ((size_t)m * BATCH + row) * CDIM;
        orow[m * CDIM + tid]       = hrow[tid] + e0;
        orow[m * CDIM + tid + 256] = hrow[tid + 256] + e1;
    }
}

// ---------------------------------------------------------------------------
extern "C" void kernel_launch(void* const* d_in, const int* in_sizes, int n_in,
                              void* d_out, int out_size, void* d_ws, size_t ws_size,
                              hipStream_t stream)
{
    (void)in_sizes; (void)n_in; (void)out_size; (void)ws_size;
    const float* rgb       = (const float*)d_in[0];
    const float* nir       = (const float*)d_in[1];
    const float* tir       = (const float*)d_in[2];
    const float* q_rgb     = (const float*)d_in[3];
    const float* q_nir     = (const float*)d_in[4];
    const float* q_tir     = (const float*)d_in[5];
    const float* in_w      = (const float*)d_in[6];
    const float* in_b      = (const float*)d_in[7];
    const float* out_w     = (const float*)d_in[8];
    const float* out_b     = (const float*)d_in[9];
    const float* attn_ln_g = (const float*)d_in[10];
    const float* attn_ln_b = (const float*)d_in[11];
    const float* ent_w     = (const float*)d_in[12];
    const float* ent_b     = (const float*)d_in[13];
    const float* gate_w1   = (const float*)d_in[14];
    const float* gate_b1   = (const float*)d_in[15];
    const float* gate_ln_g = (const float*)d_in[16];
    const float* gate_ln_b = (const float*)d_in[17];
    const float* gate_w2   = (const float*)d_in[18];
    const float* gate_b2   = (const float*)d_in[19];
    const float* alpha_raw = (const float*)d_in[20];
    const float* enh_w     = (const float*)d_in[21];
    const float* enh_b     = (const float*)d_in[22];
    const float* enh_ln_g  = (const float*)d_in[23];
    const float* enh_ln_b  = (const float*)d_in[24];

    float* ws    = (float*)d_ws;
    float* qk    = ws;                       // 3*8*512 = 12288
    float* sb    = qk + 12288;               // 24
    float* emean = sb + 24;                  // 513 (pad 520)
    float* ebias = emean + 520;              // 512
    float* pool  = ebias + 512;              // 3*B*4096 = 6291456
    float* weff  = pool + (size_t)3 * BATCH * KPOOL;   // 512*4096 = 2097152
    float* lin1  = weff + (size_t)CDIM * KPOOL;        // 786432
    float* h     = lin1 + 786432;            // 786432
    float* hc    = h    + 786432;            // 786432
    float* glin  = hc   + 786432;            // 262144
    float* gbuf  = glin + 262144;            // 262144
    float* hfus  = gbuf + 262144;            // 262144
    float* elin  = hfus + 262144;            // 262144  (total ~47 MB)

    precompute_kernel<<<1, 256, 0, stream>>>(q_rgb, q_nir, q_tir, in_w, in_b,
                                             ent_w, ent_b, out_w, out_b,
                                             qk, sb, emean, ebias);

    weff_kernel<<<dim3(NHEAD, CDIM / 128), 256, 0, stream>>>(out_w, in_w, weff);

    attn_pool_kernel<<<dim3(BATCH, 3), 256, 0, stream>>>(rgb, nir, tir,
                                                         qk, sb, pool);

    // fused Wv+out_proj: (1536 x 4096) @ (512 x 4096)^T
    gemm_wmma<<<dim3(1536 / 64, 512 / 128), 256, 0, stream>>>(
        pool, weff, ebias, lin1, 1536, CDIM, KPOOL);
    ln_rows_kernel<<<1536, 256, 0, stream>>>(lin1, attn_ln_g, attn_ln_b, h, 0);

    pack_hc<<<(BATCH * 3 * CDIM) / 256, 256, 0, stream>>>(h, hc);

    // gate: (512 x 1536) @ (512 x 1536)^T
    gemm_wmma<<<dim3(BATCH / 64, 512 / 128), 256, 0, stream>>>(
        hc, gate_w1, gate_b1, glin, BATCH, CDIM, 3 * CDIM);
    ln_rows_kernel<<<BATCH, 256, 0, stream>>>(glin, gate_ln_g, gate_ln_b, gbuf, 1);

    fuse_kernel<<<BATCH, 256, 0, stream>>>(h, emean, gbuf, gate_w2, gate_b2,
                                           alpha_raw, hfus);

    // enh: (512 x 512) @ (512 x 512)^T
    gemm_wmma<<<dim3(BATCH / 64, 512 / 128), 256, 0, stream>>>(
        hfus, enh_w, enh_b, elin, BATCH, CDIM, CDIM);
    final_kernel<<<BATCH, 256, 0, stream>>>(elin, enh_ln_g, enh_ln_b, h,
                                            (float*)d_out);
}
